// ClusteredAttention_32719060861245
// MI455X (gfx1250) — matrix operand order repaired and checked by reading, not compiled, not run on hardware
//
#include <hip/hip_runtime.h>
#include <hip/hip_bf16.h>

typedef __bf16 bf16_t;
typedef __attribute__((ext_vector_type(8)))  __bf16 v8bf;
typedef __attribute__((ext_vector_type(16))) __bf16 v16bf;
typedef __attribute__((ext_vector_type(8)))  float  v8f;
typedef __attribute__((ext_vector_type(4)))  unsigned v4u;
typedef __attribute__((ext_vector_type(4)))  int      v4i;
typedef __attribute__((ext_vector_type(8)))  int      v8i;

constexpr int N_ = 4, L_ = 4096, S_ = 4096, H_ = 8, E_ = 64, D_ = 64;
constexpr int C_ = 256, BITS_ = 32, ITERS_ = 10;
constexpr int NH_ = N_ * H_;
constexpr int SSEG_ = S_ / 4;            // S columns per wave (split-S flash)
#define SM_TEMP2  0.18033688f            /* (1/sqrt(64)) * log2(e): base-2 logits */

// ---------------------------------------------------------------------------
// DPP16 butterfly reduction across a 16-lane row (VALU only, no LDS).
// ---------------------------------------------------------------------------
template <int CTRL>
__device__ __forceinline__ float dpp_f(float v) {
    return __int_as_float(
        __builtin_amdgcn_mov_dpp(__float_as_int(v), CTRL, 0xF, 0xF, true));
}
__device__ __forceinline__ float row16_max(float v) {
    v = fmaxf(v, dpp_f<0xB1>(v));    // quad_perm(1,0,3,2)
    v = fmaxf(v, dpp_f<0x4E>(v));    // quad_perm(2,3,0,1)
    v = fmaxf(v, dpp_f<0x141>(v));   // row_half_mirror
    v = fmaxf(v, dpp_f<0x140>(v));   // row_mirror
    return v;
}
__device__ __forceinline__ float row16_sum(float v) {
    v += dpp_f<0xB1>(v);
    v += dpp_f<0x4E>(v);
    v += dpp_f<0x141>(v);
    v += dpp_f<0x140>(v);
    return v;
}

// ---------------------------------------------------------------------------
// TDM: async DMA of a 32-row x 64-col bf16 tile (row stride 64 elems) into LDS.
// ---------------------------------------------------------------------------
__device__ __forceinline__ void tdm_load_k_chunk(unsigned lds_addr, const bf16_t* gptr)
{
    unsigned long long ga = (unsigned long long)(uintptr_t)gptr;
    v4u g0;
    g0[0] = 1u;                                        // count=1 (valid user D#)
    g0[1] = lds_addr;                                  // LDS byte address
    g0[2] = (unsigned)(ga & 0xFFFFFFFFu);              // global_addr[31:0]
    g0[3] = (unsigned)((ga >> 32) & 0x1FFFFFFu)        // global_addr[56:32]
          | (2u << 30);                                // type=2 ("image")
    v8i g1;
    g1[0] = (int)(1u << 16);        // workgroup_mask=0, data_size=1 (2 bytes)
    g1[1] = (int)(64u << 16);       // tensor_dim0 = 64 (bits 79:48, low half)
    g1[2] = (int)(32u << 16);       // tensor_dim0 hi | tensor_dim1 = 32
    g1[3] = (int)(64u << 16);       // tensor_dim1 hi | tile_dim0 = 64
    g1[4] = 32;                     // tile_dim1 = 32, tile_dim2 = 0
    g1[5] = 64;                     // tensor_dim0_stride = 64 (elements)
    g1[6] = 0;
    g1[7] = 0;
    v4i z4 = {0, 0, 0, 0};
#if defined(__clang_major__) && (__clang_major__ >= 23)
    v8i z8 = {0, 0, 0, 0, 0, 0, 0, 0};
    __builtin_amdgcn_tensor_load_to_lds(g0, g1, z4, z4, z8, 0);
#else
    __builtin_amdgcn_tensor_load_to_lds(g0, g1, z4, z4, 0);
#endif
}

// ---------------------------------------------------------------------------
// Kernel 1: proj = Q @ planes[:, :E]^T + bias; pack sign bits into uint32
// ---------------------------------------------------------------------------
__global__ void __launch_bounds__(256)
k_proj_bits(const float* __restrict__ q, const float* __restrict__ planes,
            unsigned* __restrict__ bits)
{
    __shared__ float pl[BITS_ * (E_ + 1)];
    for (int i = threadIdx.x; i < BITS_ * (E_ + 1); i += blockDim.x) pl[i] = planes[i];
    __syncthreads();

    int t  = blockIdx.x * blockDim.x + threadIdx.x;   // (n*H + h)*L + l
    int l  = t % L_;
    int nh = t / L_;
    int h  = nh % H_;
    int n  = nh / H_;

    const float* qr = q + (((size_t)(n * L_ + l) * H_ + h) * E_);
    float qv[E_];
#pragma unroll
    for (int e = 0; e < E_; e += 4) {
        float4 v4 = *(const float4*)(qr + e);
        qv[e] = v4.x; qv[e + 1] = v4.y; qv[e + 2] = v4.z; qv[e + 3] = v4.w;
    }
    unsigned m = 0;
#pragma unroll 1
    for (int b = 0; b < BITS_; ++b) {
        const float* p = pl + b * (E_ + 1);
        float acc = p[E_];
#pragma unroll
        for (int e = 0; e < E_; ++e) acc = fmaf(qv[e], p[e], acc);
        if (acc > 0.0f) m |= (1u << b);
    }
    bits[(size_t)nh * L_ + l] = m;
}

// ---------------------------------------------------------------------------
// Kernel 2: Lloyd k-means in Hamming space (popcount, centroid-outer argmin).
// ---------------------------------------------------------------------------
__global__ void __launch_bounds__(256)
k_lloyd(const unsigned* __restrict__ bits, int* __restrict__ assign_g,
        unsigned* __restrict__ counts_g)
{
    __shared__ unsigned cent[C_];
    __shared__ unsigned cnt[C_];
    __shared__ unsigned sums[C_ * BITS_];   // 32 KB

    const int nh  = blockIdx.x;
    const int tid = threadIdx.x;
    const unsigned* bb = bits + (size_t)nh * L_;
    constexpr int TPT = L_ / 256;           // tokens per thread = 16

    if (tid < C_) {
        int idx = (int)(((double)tid * (double)(L_ - 1)) / (double)(C_ - 1));
        cent[tid] = bb[idx];
    }
    __syncthreads();

    unsigned myb[TPT];
#pragma unroll
    for (int i = 0; i < TPT; ++i) myb[i] = bb[tid + i * 256];

    for (int it = 0; it <= ITERS_; ++it) {
        int key[TPT];
#pragma unroll
        for (int i = 0; i < TPT; ++i) key[i] = 0x7FFFFFFF;
#pragma unroll 4
        for (int c = 0; c < C_; ++c) {
            unsigned cc = cent[c];
#pragma unroll
            for (int i = 0; i < TPT; ++i) {
                int k = (__popc(myb[i] ^ cc) << 8) | c;
                key[i] = min(key[i], k);
            }
        }
        if (it == ITERS_) {
            for (int i = tid; i < C_; i += blockDim.x) cnt[i] = 0;
            __syncthreads();
#pragma unroll
            for (int i = 0; i < TPT; ++i) {
                int bi = key[i] & 0xFF;
                assign_g[(size_t)nh * L_ + tid + i * 256] = bi;
                atomicAdd(&cnt[bi], 1u);
            }
            __syncthreads();
            if (tid < C_) counts_g[(size_t)nh * C_ + tid] = cnt[tid];
            return;
        }
        for (int i = tid; i < C_; i += blockDim.x) cnt[i] = 0;
        for (int i = tid; i < C_ * BITS_; i += blockDim.x) sums[i] = 0;
        __syncthreads();
#pragma unroll 1
        for (int i = 0; i < TPT; ++i) {
            int bi = key[i] & 0xFF;
            unsigned b = myb[i];
            atomicAdd(&cnt[bi], 1u);
            unsigned* s = &sums[bi * BITS_];
#pragma unroll 1
            for (int j = 0; j < BITS_; ++j)
                if (b & (1u << j)) atomicAdd(&s[j], 1u);
        }
        __syncthreads();
        if (tid < C_) {
            unsigned c0 = cnt[tid];
            if (c0 > 0) {
                unsigned nm = 0;
#pragma unroll
                for (int j = 0; j < BITS_; ++j)
                    if (2u * sums[tid * BITS_ + j] > c0) nm |= (1u << j);
                cent[tid] = nm;
            }
        }
        __syncthreads();
    }
}

// ---------------------------------------------------------------------------
// Kernel 3: convert K -> bf16 [nh,S,E] and V -> bf16 transposed [nh,D,S]
// ---------------------------------------------------------------------------
__global__ void __launch_bounds__(256)
k_convert_kv(const float* __restrict__ keys, const float* __restrict__ values,
             bf16_t* __restrict__ Kb, bf16_t* __restrict__ Vt)
{
    int t  = blockIdx.x * blockDim.x + threadIdx.x;  // (n*H+h)*S + s
    int s  = t % S_;
    int nh = t / S_;
    int h  = nh % H_;
    int n  = nh / H_;

    const float* kr = keys + (((size_t)(n * S_ + s) * H_ + h) * E_);
    bf16_t* kd = Kb + ((size_t)nh * S_ + s) * E_;
#pragma unroll
    for (int e = 0; e < E_; ++e) kd[e] = (bf16_t)kr[e];

    const float* vr = values + (((size_t)(n * S_ + s) * H_ + h) * D_);
    bf16_t* vd = Vt + (size_t)nh * D_ * S_ + s;
#pragma unroll
    for (int d = 0; d < D_; ++d) vd[(size_t)d * S_] = (bf16_t)vr[d];
}

// ---------------------------------------------------------------------------
// Kernel 4: cluster means Qg = scatter-mean(Q, assign) -> bf16 [nh,C,E]
// ---------------------------------------------------------------------------
__global__ void __launch_bounds__(256)
k_qg(const float* __restrict__ q, const int* __restrict__ assign_g,
     const unsigned* __restrict__ counts_g, bf16_t* __restrict__ Qg)
{
    __shared__ float qs[C_ * 32];   // 32 KB (two E-halves)
    const int nh  = blockIdx.x;
    const int h   = nh % H_;
    const int n   = nh / H_;
    const int tid = threadIdx.x;

    for (int eh = 0; eh < 2; ++eh) {
        for (int i = tid; i < C_ * 32; i += blockDim.x) qs[i] = 0.0f;
        __syncthreads();
        for (int l = tid; l < L_; l += blockDim.x) {
            int c = assign_g[(size_t)nh * L_ + l];
            const float* qr = q + (((size_t)(n * L_ + l) * H_ + h) * E_) + eh * 32;
            float* dst = &qs[c * 32];
#pragma unroll 1
            for (int e = 0; e < 32; ++e) atomicAdd(&dst[e], qr[e]);
        }
        __syncthreads();
        for (int i = tid; i < C_ * 32; i += blockDim.x) {
            int c = i >> 5;
            int e = (i & 31) + eh * 32;
            float cn = fmaxf((float)counts_g[(size_t)nh * C_ + c], 1.0f);
            Qg[((size_t)nh * C_ + c) * E_ + e] = (bf16_t)(qs[i] / cn);
        }
        __syncthreads();
    }
}

// ---------------------------------------------------------------------------
// Kernel 5: clustered flash attention, WMMA bf16 + TDM-pipelined K chunks,
// split-S across 4 waves per block with log-sum-exp merge through LDS.
// Block = one (n,h,ctile); wave w processes S columns [w*1024, (w+1)*1024).
// ---------------------------------------------------------------------------
__global__ void __launch_bounds__(128)
k_attn(const bf16_t* __restrict__ Qg, const bf16_t* __restrict__ Kb,
       const bf16_t* __restrict__ Vt, float* __restrict__ Vc)
{
    __shared__ bf16_t ldsP[4][16 * 32];        // per-wave P staging, 4 KB
    __shared__ bf16_t ldsK[4][2][32 * 64];     // per-wave double K buffer, 32 KB
    __shared__ float  smax[4][16];             // per-wave row maxima
    __shared__ float  ssum[4][16];             // per-wave row sums
    __shared__ float  Oacc[16 * 64];           // merged output tile, 4 KB
    __shared__ float  gsum[16];                // merged row sums

    const int tid  = threadIdx.x;
    const int wave = tid >> 5;
    const int lane = tid & 31;
    const int nh   = blockIdx.x >> 4;
    const int ct   = blockIdx.x & 15;
    const int m    = lane & 15;                // row/col index within tile
    const int half = lane >> 4;
    const int s0   = wave * SSEG_;             // this wave's S segment base

    // Zero merge accumulators (visible to all by the post-loop barrier).
    for (int i = tid; i < 16 * 64; i += 128) Oacc[i] = 0.0f;
    if (tid < 16) gsum[tid] = 0.0f;

    // ---- A fragments (Qg rows, contiguous k = 16*half + s) -----------------
    const bf16_t* qrow = Qg + ((size_t)nh * C_ + (ct * 16 + m)) * E_;
    v8bf a00 = *(const v8bf*)(qrow +      half * 16);
    v8bf a01 = *(const v8bf*)(qrow +  8 + half * 16);
    v8bf a10 = *(const v8bf*)(qrow + 32 + half * 16);
    v8bf a11 = *(const v8bf*)(qrow + 40 + half * 16);
    v16bf aq0 = __builtin_shufflevector(a00, a01, 0,1,2,3,4,5,6,7,8,9,10,11,12,13,14,15);
    v16bf aq1 = __builtin_shufflevector(a10, a11, 0,1,2,3,4,5,6,7,8,9,10,11,12,13,14,15);

    v8f o0 = {}, o1 = {}, o2 = {}, o3 = {};
    float rmax[8], rsum[8];
#pragma unroll
    for (int r = 0; r < 8; ++r) { rmax[r] = -3.0e38f; rsum[r] = 0.0f; }

    bf16_t* P = &ldsP[wave][0];
    const bf16_t* Kbase = Kb + (size_t)nh * S_ * E_;
    const bf16_t* Vbase = Vt + (size_t)nh * D_ * S_;
    const unsigned kaddr[2] = {
        (unsigned)(uintptr_t)&ldsK[wave][0][0],
        (unsigned)(uintptr_t)&ldsK[wave][1][0] };

    // Prologue: DMA this segment's chunk 0 into buffer 0.
    tdm_load_k_chunk(kaddr[0], Kbase + (size_t)s0 * E_);

    for (int it = 0; it < SSEG_ / 32; ++it) {
        const int sb  = s0 + it * 32;
        const int cur = it & 1;

        // DMA next chunk (wrap on the last iter; buffer not in use).
        tdm_load_k_chunk(kaddr[cur ^ 1],
                         Kbase + (size_t)((sb + 32) & (S_ - 1)) * E_);

        // Issue V-fragment loads early; latency hides under QK + softmax.
        const size_t vsoff = (size_t)sb + half * 16;
        v16bf bv0 = *(const v16bf*)(Vbase + (size_t)( 0 + m) * S_ + vsoff);
        v16bf bv1 = *(const v16bf*)(Vbase + (size_t)(16 + m) * S_ + vsoff);
        v16bf bv2 = *(const v16bf*)(Vbase + (size_t)(32 + m) * S_ + vsoff);
        v16bf bv3 = *(const v16bf*)(Vbase + (size_t)(48 + m) * S_ + vsoff);
        __builtin_prefetch(Vbase + (size_t)m * S_ + ((sb + 32) & (S_ - 1)), 0, 1);

        // Wait for the *previous* TDM (<=1 outstanding), then read K frags.
        __builtin_amdgcn_s_wait_tensorcnt(1);
        asm volatile("" ::: "memory");
        const bf16_t* kc = &ldsK[wave][cur][0];
        v16bf b0 = *(const v16bf*)(kc + (size_t)( 0 + m) * E_ +      half * 16);
        v16bf b1 = *(const v16bf*)(kc + (size_t)( 0 + m) * E_ + 32 + half * 16);
        v16bf b2 = *(const v16bf*)(kc + (size_t)(16 + m) * E_ +      half * 16);
        v16bf b3 = *(const v16bf*)(kc + (size_t)(16 + m) * E_ + 32 + half * 16);

        // ---- QK for two 16-column tiles (4 WMMAs) -------------------------
        v8f qkA = {}, qkB = {};
        qkA = __builtin_amdgcn_wmma_f32_16x16x32_bf16(false, aq0, false, b0,
                                                      (short)0, qkA, false, false);
        qkA = __builtin_amdgcn_wmma_f32_16x16x32_bf16(false, aq1, false, b1,
                                                      (short)0, qkA, false, false);
        qkB = __builtin_amdgcn_wmma_f32_16x16x32_bf16(false, aq0, false, b2,
                                                      (short)0, qkB, false, false);
        qkB = __builtin_amdgcn_wmma_f32_16x16x32_bf16(false, aq1, false, b3,
                                                      (short)0, qkB, false, false);

        // ---- online softmax (base-2), rows = r + 8*half, DPP16 reduce -----
        float f[8];
#pragma unroll
        for (int r = 0; r < 8; ++r) {
            float zA = qkA[r] * SM_TEMP2;            // base-2 logits
            float zB = qkB[r] * SM_TEMP2;
            float tm = row16_max(fmaxf(zA, zB));
            float nm = fmaxf(rmax[r], tm);
            f[r] = exp2f(rmax[r] - nm);
            rmax[r] = nm;
            float pA = exp2f(zA - nm);
            float pB = exp2f(zB - nm);
            float ts = row16_sum(pA + pB);
            rsum[r] = rsum[r] * f[r] + ts;

            int row = r + half * 8;
            P[row * 32 + m]      = (bf16_t)pA;       // C-layout -> LDS tile
            P[row * 32 + 16 + m] = (bf16_t)pB;
        }
#pragma unroll
        for (int r = 0; r < 8; ++r) {
            o0[r] *= f[r]; o1[r] *= f[r]; o2[r] *= f[r]; o3[r] *= f[r];
        }

        // wave-internal LDS RAW: all lanes' P stores must land before reads
        asm volatile("s_wait_dscnt 0" ::: "memory");

        // ---- read P back as A fragment (16x32, bf16) ----------------------
        v8bf plo = *(const v8bf*)(P + m * 32 +     half * 16);
        v8bf phi = *(const v8bf*)(P + m * 32 + 8 + half * 16);
        v16bf ap = __builtin_shufflevector(plo, phi,
                       0,1,2,3,4,5,6,7,8,9,10,11,12,13,14,15);

        // ---- P x V: 4 WMMAs (d tiles of 16) -------------------------------
        o0 = __builtin_amdgcn_wmma_f32_16x16x32_bf16(false, ap, false, bv0,
                                                     (short)0, o0, false, false);
        o1 = __builtin_amdgcn_wmma_f32_16x16x32_bf16(false, ap, false, bv1,
                                                     (short)0, o1, false, false);
        o2 = __builtin_amdgcn_wmma_f32_16x16x32_bf16(false, ap, false, bv2,
                                                     (short)0, o2, false, false);
        o3 = __builtin_amdgcn_wmma_f32_16x16x32_bf16(false, ap, false, bv3,
                                                     (short)0, o3, false, false);
    }

    // ---- split-S merge: log-sum-exp combine of 4 segment partials ---------
    if (m == 0) {                     // one lane per (row, half) pair
#pragma unroll
        for (int r = 0; r < 8; ++r) {
            smax[wave][r + half * 8] = rmax[r];
            ssum[wave][r + half * 8] = rsum[r];
        }
    }
    __syncthreads();

#pragma unroll
    for (int r = 0; r < 8; ++r) {
        const int row = r + half * 8;
        float gm = fmaxf(fmaxf(smax[0][row], smax[1][row]),
                         fmaxf(smax[2][row], smax[3][row]));
        float w = exp2f(rmax[r] - gm);           // this wave's weight for row
        float* od = &Oacc[row * 64];
        atomicAdd(&od[ 0 + m], o0[r] * w);
        atomicAdd(&od[16 + m], o1[r] * w);
        atomicAdd(&od[32 + m], o2[r] * w);
        atomicAdd(&od[48 + m], o3[r] * w);
        if (m == 0) atomicAdd(&gsum[row], rsum[r] * w);
    }
    __syncthreads();

    // ---- normalize and store Vc [nh,C,D] f32 ------------------------------
    for (int i = tid; i < 16 * 64; i += 128) {
        int row = i >> 6;
        int d   = i & 63;
        int c   = ct * 16 + row;
        Vc[((size_t)nh * C_ + c) * D_ + d] = Oacc[i] / gsum[row];
    }
}

// ---------------------------------------------------------------------------
// Kernel 6: broadcast cluster outputs back to tokens; out [N,L,H,D] f32
// ---------------------------------------------------------------------------
__global__ void __launch_bounds__(256)
k_gather(const float* __restrict__ Vc, const int* __restrict__ assign_g,
         float* __restrict__ out)
{
    int t  = blockIdx.x * blockDim.x + threadIdx.x;  // (n*L + l)*H + h
    int h  = t % H_;
    int nl = t / H_;
    int l  = nl % L_;
    int n  = nl / L_;
    int nh = n * H_ + h;
    int c  = assign_g[(size_t)nh * L_ + l];
    const float* src = Vc + ((size_t)nh * C_ + c) * D_;
    float* dst = out + (size_t)t * D_;
#pragma unroll
    for (int d = 0; d < D_; d += 4)
        *(float4*)(dst + d) = *(const float4*)(src + d);
}

// ---------------------------------------------------------------------------
extern "C" void kernel_launch(void* const* d_in, const int* in_sizes, int n_in,
                              void* d_out, int out_size, void* d_ws, size_t ws_size,
                              hipStream_t stream)
{
    (void)in_sizes; (void)n_in; (void)out_size; (void)ws_size;
    const float* q      = (const float*)d_in[0];
    const float* k      = (const float*)d_in[1];
    const float* v      = (const float*)d_in[2];
    const float* planes = (const float*)d_in[3];

    char* ws = (char*)d_ws;
    size_t off = 0;
    auto alloc = [&](size_t bytes) -> void* {
        void* p = ws + off;
        off += (bytes + 255) & ~(size_t)255;
        return p;
    };
    unsigned* bits   = (unsigned*)alloc((size_t)NH_ * L_ * sizeof(unsigned));
    int*      assign = (int*)     alloc((size_t)NH_ * L_ * sizeof(int));
    unsigned* counts = (unsigned*)alloc((size_t)NH_ * C_ * sizeof(unsigned));
    bf16_t*   Qg     = (bf16_t*)  alloc((size_t)NH_ * C_ * E_ * sizeof(bf16_t));
    bf16_t*   Kb     = (bf16_t*)  alloc((size_t)NH_ * S_ * E_ * sizeof(bf16_t));
    bf16_t*   Vt     = (bf16_t*)  alloc((size_t)NH_ * D_ * S_ * sizeof(bf16_t));
    float*    Vc     = (float*)   alloc((size_t)NH_ * C_ * D_ * sizeof(float));

    k_proj_bits <<<(NH_ * L_) / 256, 256, 0, stream>>>(q, planes, bits);
    k_convert_kv<<<(NH_ * S_) / 256, 256, 0, stream>>>(k, v, Kb, Vt);
    k_lloyd     <<<NH_,              256, 0, stream>>>(bits, assign, counts);
    k_qg        <<<NH_,              256, 0, stream>>>(q, assign, counts, Qg);
    k_attn      <<<NH_ * (C_ / 16),  128, 0, stream>>>(Qg, Kb, Vt, Vc);
    k_gather    <<<(N_ * L_ * H_) / 256, 256, 0, stream>>>(Vc, assign, (float*)d_out);
}
